// Attention_66907000537276
// MI455X (gfx1250) — matrix operand order repaired and checked by reading, not compiled, not run on hardware
//
#include <hip/hip_runtime.h>
#include <hip/hip_bf16.h>

typedef _Float16 half_t;
typedef __attribute__((ext_vector_type(16))) _Float16 v16h;
typedef __attribute__((ext_vector_type(8)))  _Float16 v8h;
typedef __attribute__((ext_vector_type(8)))  float    v8f;
typedef __attribute__((ext_vector_type(4)))  float    v4f;
typedef __attribute__((ext_vector_type(4)))  int      v4i;

// ---------------------------------------------------------------------------
// CDNA5 async global->LDS copy (ASYNCcnt-tracked), with portable fallback.
// Probe result: builtin signature is (v4i addrspace(1)* src, lds dst, Ii, Ii).
// ---------------------------------------------------------------------------
#ifdef __has_builtin
#  if __has_builtin(__builtin_amdgcn_global_load_async_to_lds_b128)
#    define HAVE_ASYNC_LDS 1
#  endif
#endif
#ifndef HAVE_ASYNC_LDS
#  define HAVE_ASYNC_LDS 0
#endif

#if HAVE_ASYNC_LDS
typedef __attribute__((address_space(1))) v4i gv4i;
typedef __attribute__((address_space(3))) v4i lv4i;
__device__ inline void async_cp_b128(const half_t* gsrc, half_t* ldst) {
    __builtin_amdgcn_global_load_async_to_lds_b128((gv4i*)gsrc, (lv4i*)ldst,
                                                   /*offset=*/0, /*cpol=*/0);
}
__device__ inline void async_join() {
#  if __has_builtin(__builtin_amdgcn_s_wait_asynccnt)
    __builtin_amdgcn_s_wait_asynccnt(0);
#  else
    asm volatile("s_wait_asynccnt 0x0" ::: "memory");
#  endif
}
#else
__device__ inline void async_cp_b128(const half_t* gsrc, half_t* ldst) {
    *(v8h*)ldst = *(const v8h*)gsrc;
}
__device__ inline void async_join() {}
#endif

// ---------------------------------------------------------------------------
// WMMA fragment helpers (wave32, V_WMMA_F32_16X16X32_F16)
//   A (16x32): lane L -> row L%16; elems j=0..7 -> K=8*(L/16)+j, j=8..15 -> +16
//   B (32x16): lane L -> N=L%16; elems j=0..7 -> K=8*(L/16)+j, j=8..15 -> +16
//   C/D:       vgpr r, lane L -> M = r + 8*(L/16), N = L%16
// ---------------------------------------------------------------------------
__device__ inline v16h load_frag_a(const half_t* lds, int ld, int row0, int k0, int lane) {
    const int m = lane & 15;
    const int g = lane >> 4;
    const half_t* p = lds + (row0 + m) * ld + k0 + 8 * g;
    v8h lo = *(const v8h*)p;
    v8h hi = *(const v8h*)(p + 16);
    v16h r;
#pragma unroll
    for (int i = 0; i < 8; ++i) { r[i] = lo[i]; r[8 + i] = hi[i]; }
    return r;
}

__device__ inline v16h load_frag_b(const half_t* lds, int ld, int k0, int n0, int lane) {
    const int n = lane & 15;
    const int g = lane >> 4;
    const half_t* p = lds + (k0 + 8 * g) * ld + n0 + n;
    v16h r;
#pragma unroll
    for (int i = 0; i < 8; ++i) { r[i] = p[i * ld]; r[8 + i] = p[(16 + i) * ld]; }
    return r;
}

__device__ inline v8f wmma_f16(v16h a, v16h b, v8f c) {
    return __builtin_amdgcn_wmma_f32_16x16x32_f16(false, a, false, b, (short)0, c,
                                                  false, false);
}

// ---------------------------------------------------------------------------
// Problem constants
// ---------------------------------------------------------------------------
#define BATCH 16
#define CCH   192
#define HH    128
#define WW    128
#define NPIX  (HH * WW)          // 16384
#define HEADS 4
#define HDIM  48
#define C3    (3 * CCH)          // 576

// ---------------------------------------------------------------------------
// Kernel: zero scratch
// ---------------------------------------------------------------------------
__global__ void zero_kernel(float* __restrict__ p, int n) {
    for (int i = blockIdx.x * blockDim.x + threadIdx.x; i < n;
         i += gridDim.x * blockDim.x)
        p[i] = 0.0f;
}

// ---------------------------------------------------------------------------
// Kernel: fp32 -> fp16 conversion (8 elems / thread / iter, b128 stores)
// ---------------------------------------------------------------------------
__global__ __launch_bounds__(256) void cvt_kernel(const float* __restrict__ src,
                                                  half_t* __restrict__ dst, long n) {
    const long stride = (long)gridDim.x * blockDim.x * 8;
    for (long i = ((long)blockIdx.x * blockDim.x + threadIdx.x) * 8; i < n; i += stride) {
        v4f a = *(const v4f*)(src + i);
        v4f b = *(const v4f*)(src + i + 4);
        v8h o;
#pragma unroll
        for (int j = 0; j < 4; ++j) { o[j] = (half_t)a[j]; o[4 + j] = (half_t)b[j]; }
        *(v8h*)(dst + i) = o;
    }
}

// ---------------------------------------------------------------------------
// Kernel: fp16 WMMA GEMM   C[b] = A(MxK) @ B[b](KxN)
//   Block tile 64(M) x 64(N), BK=32, 8 waves (4x2), 2 WMMA tiles per wave.
//   Both tiles staged via GLOBAL_LOAD_ASYNC_TO_LDS_B128.
// ---------------------------------------------------------------------------
template <bool OUT_F32>
__global__ __launch_bounds__(256) void gemm_h_kernel(
    const half_t* __restrict__ A, const half_t* __restrict__ B,
    void* __restrict__ Cv, int M, int K, int N, long bStride, long cStride) {
    __shared__ __align__(16) half_t As[64 * 40];
    __shared__ __align__(16) half_t Bs[32 * 72];

    const int t = threadIdx.x;
    const int lane = t & 31;
    const int wave = t >> 5;
    const int wm = wave >> 1;  // 0..3
    const int wn = wave & 1;   // 0..1
    const int m0 = blockIdx.y * 64;
    const int n0 = blockIdx.x * 64;
    const half_t* Bb = B + (long)blockIdx.z * bStride;

    // per-thread staging coordinates
    const int ar = t >> 2, ac = (t & 3) * 8;   // A: 64 rows x 4 chunks
    const int br = t >> 3, bc = (t & 7) * 8;   // B: 32 rows x 8 chunks

    v8f acc0 = {}, acc1 = {};
    for (int k0 = 0; k0 < K; k0 += 32) {
        __syncthreads();
        async_cp_b128(A + (long)(m0 + ar) * K + k0 + ac, &As[ar * 40 + ac]);
        async_cp_b128(Bb + (long)(k0 + br) * N + n0 + bc, &Bs[br * 72 + bc]);
        async_join();
        __syncthreads();
        v16h a  = load_frag_a(As, 40, wm * 16, 0, lane);
        v16h b0 = load_frag_b(Bs, 72, 0, wn * 32 + 0, lane);
        v16h b1 = load_frag_b(Bs, 72, 0, wn * 32 + 16, lane);
        acc0 = wmma_f16(a, b0, acc0);
        acc1 = wmma_f16(a, b1, acc1);
    }

    const int g = lane >> 4;
    const int col0 = n0 + wn * 32 + (lane & 15);
#pragma unroll
    for (int r = 0; r < 8; ++r) {
        const int row = m0 + wm * 16 + r + 8 * g;
        const long idx = (long)blockIdx.z * cStride + (long)row * N + col0;
        if (OUT_F32) {
            ((float*)Cv)[idx] = acc0[r];
            ((float*)Cv)[idx + 16] = acc1[r];
        } else {
            ((half_t*)Cv)[idx] = (half_t)acc0[r];
            ((half_t*)Cv)[idx + 16] = (half_t)acc1[r];
        }
    }
}

// ---------------------------------------------------------------------------
// Kernel: depthwise 3x3 (SAME, groups=3C) + sum-of-squares for q/k rows.
// ---------------------------------------------------------------------------
__global__ __launch_bounds__(256) void dwconv_kernel(
    const half_t* __restrict__ qkv, const float* __restrict__ wdw,
    half_t* __restrict__ dst, float* __restrict__ qn2, float* __restrict__ kn2) {
    const int bo = blockIdx.x;            // b*576 + o
    const int b = bo / C3, o = bo % C3;

    float w9[9];
#pragma unroll
    for (int i = 0; i < 9; ++i) w9[i] = wdw[o * 9 + i];

    const half_t* src = qkv + (long)bo * NPIX;
    half_t* out = dst + (long)bo * NPIX;

    float ss = 0.0f;
    for (int p = threadIdx.x; p < NPIX; p += 256) {
        const int h = p >> 7, w = p & 127;
        float acc = 0.0f;
#pragma unroll
        for (int dh = -1; dh <= 1; ++dh) {
            const int hh = h + dh;
            if (hh < 0 || hh > HH - 1) continue;
#pragma unroll
            for (int dw = -1; dw <= 1; ++dw) {
                const int ww = w + dw;
                if (ww < 0 || ww > WW - 1) continue;
                acc += w9[(dh + 1) * 3 + (dw + 1)] * (float)src[hh * WW + ww];
            }
        }
        out[p] = (half_t)acc;
        if (o < 2 * CCH) ss += acc * acc;
    }

    if (o < 2 * CCH) {
#pragma unroll
        for (int off = 16; off > 0; off >>= 1) ss += __shfl_down(ss, off, 32);
        if ((threadIdx.x & 31) == 0) {
            float* tgt = (o < CCH) ? &qn2[b * CCH + o] : &kn2[b * CCH + (o - CCH)];
            atomicAdd(tgt, ss);
        }
    }
}

// ---------------------------------------------------------------------------
// Kernel: S[b,h] += Q(48xN) @ K(48xN)^T   (raw logits)
//   9 waves, one 16x16 tile each; positions split across gridDim.x blocks.
//   Q staged via async b128; K tile transposed via v8h loads + b16 scatters.
// ---------------------------------------------------------------------------
__global__ __launch_bounds__(288) void qk_kernel(
    const half_t* __restrict__ dw, float* __restrict__ S) {
    const int bh = blockIdx.y;
    const int b = bh >> 2, h = bh & 3;
    const int t = threadIdx.x;
    const int wave = t >> 5;            // 0..8
    const int lane = t & 31;
    const int mi = wave / 3, ni = wave % 3;

    __shared__ __align__(16) half_t Qs[48 * 40];   // [c][pos]
    __shared__ __align__(16) half_t KT[32 * 56];   // [pos][c]

    const long qbase = ((long)b * C3 + h * HDIM) * NPIX;
    const long kbase = ((long)b * C3 + CCH + h * HDIM) * NPIX;

    const int chunksPerBlock = (NPIX / 32) / gridDim.x;
    const int chunk0 = blockIdx.x * chunksPerBlock;

    v8f acc = {};
    for (int ci = 0; ci < chunksPerBlock; ++ci) {
        const int kpos = (chunk0 + ci) * 32;
        __syncthreads();
        if (t < 192) {                          // Q: 48 rows x 4 b128 chunks
            const int r = t >> 2, c = (t & 3) * 8;
            async_cp_b128(dw + qbase + (long)r * NPIX + kpos + c, &Qs[r * 40 + c]);
        }
        if (t >= 96) {                          // K^T: 48 rows x 4 v8h chunks
            const int u = t - 96;
            const int r = u >> 2, c = (u & 3) * 8;
            v8h v = *(const v8h*)(dw + kbase + (long)r * NPIX + kpos + c);
#pragma unroll
            for (int j = 0; j < 8; ++j) KT[(c + j) * 56 + r] = v[j];
        }
        async_join();
        __syncthreads();
        v16h a = load_frag_a(Qs, 40, mi * 16, 0, lane);
        v16h bb = load_frag_b(KT, 56, 0, ni * 16, lane);
        acc = wmma_f16(a, bb, acc);
    }

    const int g = lane >> 4;
    const int col = ni * 16 + (lane & 15);
#pragma unroll
    for (int r = 0; r < 8; ++r) {
        const int row = mi * 16 + r + 8 * g;
        atomicAdd(&S[(long)bh * (HDIM * HDIM) + row * HDIM + col], acc[r]);
    }
}

// ---------------------------------------------------------------------------
// Kernel: fold L2 norms + temperature, row softmax, fp16 out (48x64 padded)
// ---------------------------------------------------------------------------
__global__ __launch_bounds__(64) void softmax_kernel(
    const float* __restrict__ S, const float* __restrict__ qn2,
    const float* __restrict__ kn2, const float* __restrict__ temp,
    half_t* __restrict__ Aout) {
    const int bh = blockIdx.x;
    const int h = bh & 3;
    const int i = threadIdx.x;
    if (i >= HDIM) return;

    const float tmp = temp[h];
    const float qi = 1.0f / fmaxf(sqrtf(qn2[bh * HDIM + i]), 1e-12f);

    float row[HDIM];
    float mx = -3.0e38f;
#pragma unroll 4
    for (int j = 0; j < HDIM; ++j) {
        const float kj = 1.0f / fmaxf(sqrtf(kn2[bh * HDIM + j]), 1e-12f);
        const float v = S[(long)bh * (HDIM * HDIM) + i * HDIM + j] * qi * kj * tmp;
        row[j] = v;
        mx = fmaxf(mx, v);
    }
    float sum = 0.0f;
#pragma unroll 4
    for (int j = 0; j < HDIM; ++j) {
        row[j] = __expf(row[j] - mx);
        sum += row[j];
    }
    const float inv = 1.0f / sum;
    half_t* dst = Aout + (long)bh * (HDIM * 64) + i * 64;
#pragma unroll 4
    for (int j = 0; j < 64; ++j)
        dst[j] = (j < HDIM) ? (half_t)(row[j] * inv) : (half_t)0.0f;
}

// ---------------------------------------------------------------------------
// Kernel: out[b,h] = A(48x48) @ V(48xN); A staged once via async b128.
// ---------------------------------------------------------------------------
__global__ __launch_bounds__(256) void av_kernel(
    const half_t* __restrict__ Abuf, const half_t* __restrict__ dw,
    half_t* __restrict__ attn_out) {
    const int bh = blockIdx.y;
    const int b = bh >> 2, h = bh & 3;
    const int wave = threadIdx.x >> 5;   // 0..7 -> n tile
    const int lane = threadIdx.x & 31;

    __shared__ __align__(16) half_t Ash[48 * 72];
    for (int i = threadIdx.x; i < 384; i += 256) {        // 48 rows x 8 b128
        const int r = i >> 3, c = (i & 7) * 8;
        async_cp_b128(Abuf + (long)bh * (HDIM * 64) + r * 64 + c, &Ash[r * 72 + c]);
    }
    async_join();
    __syncthreads();

    const long vbase = ((long)b * C3 + 2 * CCH + h * HDIM) * NPIX;
    const int n0 = blockIdx.x * 128 + wave * 16;

    for (int mi = 0; mi < 3; ++mi) {
        v8f acc = {};
#pragma unroll
        for (int kk = 0; kk < 64; kk += 32) {
            v16h a = load_frag_a(Ash, 72, mi * 16, kk, lane);
            const half_t* p = dw + vbase + n0 + (lane & 15);
            v16h bb;
#pragma unroll
            for (int q = 0; q < 8; ++q) {
                int c0 = kk + 8 * (lane >> 4) + q;
                int c1 = c0 + 16;
                if (c0 > HDIM - 1) c0 = HDIM - 1;   // padded K cols of A are zero
                if (c1 > HDIM - 1) c1 = HDIM - 1;
                bb[q] = p[(long)c0 * NPIX];
                bb[8 + q] = p[(long)c1 * NPIX];
            }
            acc = wmma_f16(a, bb, acc);
        }
        const int g = lane >> 4;
        const int col = n0 + (lane & 15);
#pragma unroll
        for (int r = 0; r < 8; ++r) {
            const int row = mi * 16 + r + 8 * g;
            attn_out[((long)b * CCH + h * HDIM + row) * NPIX + col] = (half_t)acc[r];
        }
    }
}

// ---------------------------------------------------------------------------
// Launcher
// ---------------------------------------------------------------------------
extern "C" void kernel_launch(void* const* d_in, const int* in_sizes, int n_in,
                              void* d_out, int out_size, void* d_ws, size_t ws_size,
                              hipStream_t stream) {
    const float* x      = (const float*)d_in[0];   // (16,192,128,128)
    const float* w_qkv  = (const float*)d_in[1];   // (576,192,1,1)
    const float* w_dw   = (const float*)d_in[2];   // (576,1,3,3)
    const float* w_proj = (const float*)d_in[3];   // (192,192,1,1)
    const float* temp   = (const float*)d_in[4];   // (4,1,1)
    float* out = (float*)d_out;

    char* ws = (char*)d_ws;
    const long qkvBytes = (long)BATCH * C3 * NPIX * 2;        // 302 MB fp16
    const long off_qkv = 0;                                    // reused as attn_out
    const long off_dw  = off_qkv + qkvBytes;                   // also holds xh early
    const long off_S   = off_dw + qkvBytes;
    const long off_qn2 = off_S + (long)BATCH * HEADS * HDIM * HDIM * 4;
    const long off_kn2 = off_qn2 + (long)BATCH * CCH * 4;
    const long off_A   = off_kn2 + (long)BATCH * CCH * 4;
    const long off_wh  = off_A + (long)BATCH * HEADS * HDIM * 64 * 2;

    half_t* qkv      = (half_t*)(ws + off_qkv);
    half_t* dwout    = (half_t*)(ws + off_dw);
    half_t* xh       = (half_t*)(ws + off_dw);   // overlays dwout (dead before dwconv)
    float*  S        = (float*)(ws + off_S);
    float*  qn2      = (float*)(ws + off_qn2);
    float*  kn2      = (float*)(ws + off_kn2);
    half_t* Abuf     = (half_t*)(ws + off_A);
    half_t* whq      = (half_t*)(ws + off_wh);               // 576x192 fp16
    half_t* whp      = whq + (long)C3 * CCH;                 // 192x192 fp16
    half_t* attn_out = (half_t*)(ws + off_qkv);  // overlays dead qkv buffer

    // 0) zero S + qn2 + kn2 (contiguous)
    const int zn = BATCH * HEADS * HDIM * HDIM + 2 * BATCH * CCH;
    zero_kernel<<<(zn + 255) / 256, 256, 0, stream>>>(S, zn);

    // 1) one-time fp32 -> fp16 conversions (x and both weight matrices)
    const long nx = (long)BATCH * CCH * NPIX;
    cvt_kernel<<<4096, 256, 0, stream>>>(x, xh, nx);
    cvt_kernel<<<64, 256, 0, stream>>>(w_qkv, whq, (long)C3 * CCH);
    cvt_kernel<<<32, 256, 0, stream>>>(w_proj, whp, (long)CCH * CCH);

    // 2) qkv = Wqkv(576x192) @ X[b](192x16384) -> fp16
    gemm_h_kernel<false><<<dim3(NPIX / 64, C3 / 64, BATCH), 256, 0, stream>>>(
        whq, xh, qkv, C3, CCH, NPIX, (long)CCH * NPIX, (long)C3 * NPIX);

    // 3) depthwise 3x3 + sum-of-squares for q/k rows (overwrites xh region)
    dwconv_kernel<<<BATCH * C3, 256, 0, stream>>>(qkv, w_dw, dwout, qn2, kn2);

    // 4) raw logits S[b,h] = Q @ K^T (positions split across 8 blocks)
    qk_kernel<<<dim3(8, BATCH * HEADS), 288, 0, stream>>>(dwout, S);

    // 5) norm folding + softmax -> fp16 attention (48x64 zero-padded)
    softmax_kernel<<<BATCH * HEADS, 64, 0, stream>>>(S, qn2, kn2, temp, Abuf);

    // 6) attn @ V -> fp16 (b,192,N), overlays dead qkv buffer
    av_kernel<<<dim3(NPIX / 128, BATCH * HEADS), 256, 0, stream>>>(Abuf, dwout, attn_out);

    // 7) out = Wproj(192x192) @ attn_out[b](192x16384) -> fp32 d_out
    gemm_h_kernel<true><<<dim3(NPIX / 64, CCH / 64, BATCH), 256, 0, stream>>>(
        whp, attn_out, out, CCH, CCH, NPIX, (long)CCH * NPIX, (long)CCH * NPIX);
}
